// OwlWrapper_70935679861400
// MI455X (gfx1250) — compile-verified, matrix operand order checked by reading, not run on hardware
//
#include <hip/hip_runtime.h>
#include <hip/hip_bf16.h>
#include <stdint.h>

// ---------------- constants ----------------
#define DMODEL 768
#define IMLP   3072
#define NLAYER 12
#define NHEAD  12
#define HDIM   64
#define NPATCH 2304      // 48*48
#define LTOK   2305      // cls + patches
#define NTP    2336      // tokens padded to multiple of 32
#define NQ     1203
#define QD     512

#define ACT_NONE      0
#define ACT_QUICKGELU 1
#define ACT_GELU      2

#define OUT_F32   0
#define OUT_BF16  1
#define OUT_BF16T 2      // store C^T (ld = M) in bf16

typedef __attribute__((ext_vector_type(16))) __bf16 v16bf;
typedef __attribute__((ext_vector_type(8)))  float  v8f;

static inline int ceil_div_i(int a, int b) { return (a + b - 1) / b; }

// ---------------- WMMA fragment helper (bf16 operands) ----------------
// ISA 7.12.2 16-bit A/B layout for 16x32 (MxK):
//   lane L (0..15): row M=L, K = {k0..k0+7} in f[0..7], {k0+16..k0+23} in f[8..15]
//   lane L+16:      row M=L, same pattern with k0+8
// Two contiguous 16-byte chunks per lane -> b128 loads. Works for global and LDS pointers.
__device__ __forceinline__ v16bf frag_rowK_bf16(const __bf16* __restrict__ base, int ld,
                                                int row, int k0, int lane) {
  const __bf16* p = base + (size_t)row * (size_t)ld + k0 + ((lane & 16) >> 1);
  v16bf f;
#pragma unroll
  for (int j = 0; j < 8; ++j) f[j] = p[j];
#pragma unroll
  for (int j = 0; j < 8; ++j) f[8 + j] = p[16 + j];
  return f;
}

// ---------------- generic fused GEMM ----------------
// C = act((A @ W^T + bias) * scale) [+ res]; A:[M,K] bf16, W:[N,K] bf16
// One wave computes a 32x64 tile: 8 independent accumulators hide WMMA RAW hazards;
// branchless K-loop (OOB columns clamped to N-1; their results are never stored).
__global__ void __launch_bounds__(256)
gemm_wmma_kernel(const __bf16* __restrict__ A, const __bf16* __restrict__ W,
                 const float* __restrict__ bias, const float* __restrict__ res,
                 void* __restrict__ Cout, int M, int N, int K,
                 float post_scale, int act, int out_mode) {
  const int lane = threadIdx.x & 31;
  const int wave = threadIdx.x >> 5;
  const int tilesN = (N + 63) >> 6;
  const int tilesM = M >> 5;               // M is a multiple of 32 in all calls
  const int tile = blockIdx.x * 8 + wave;
  if (tile >= tilesM * tilesN) return;
  const int tm = tile / tilesN;
  const int tn = tile - tm * tilesN;
  const int l15 = lane & 15;
  const int half8 = (lane & 16) >> 1;

  const int ar0 = tm * 32 + l15;
  const int ar1 = ar0 + 16;
  int nc[4];
#pragma unroll
  for (int c = 0; c < 4; ++c) {
    int col = tn * 64 + c * 16 + l15;
    nc[c] = (col < N) ? col : (N - 1);     // clamp: garbage columns are never stored
  }

  v8f acc[2][4];
#pragma unroll
  for (int i = 0; i < 2; ++i)
#pragma unroll
    for (int c = 0; c < 4; ++c) acc[i][c] = (v8f){};

  for (int k0 = 0; k0 < K; k0 += 32) {
    if (k0 + 32 < K) {
      __builtin_prefetch((const void*)(A + (size_t)ar0 * K + k0 + 32));   // global_prefetch_b8
      __builtin_prefetch((const void*)(W + (size_t)nc[0] * K + k0 + 32));
    }
    v16bf a0 = frag_rowK_bf16(A, K, ar0, k0, lane);
    v16bf a1 = frag_rowK_bf16(A, K, ar1, k0, lane);
#pragma unroll
    for (int c = 0; c < 4; ++c) {
      v16bf b = frag_rowK_bf16(W, K, nc[c], k0, lane);
      acc[0][c] = __builtin_amdgcn_wmma_f32_16x16x32_bf16(false, a0, false, b, (short)0, acc[0][c], false, false);
      acc[1][c] = __builtin_amdgcn_wmma_f32_16x16x32_bf16(false, a1, false, b, (short)0, acc[1][c], false, false);
    }
  }

  float* Cf = (float*)Cout;
  __bf16* Cb = (__bf16*)Cout;
#pragma unroll
  for (int c = 0; c < 4; ++c) {
    const int col = tn * 64 + c * 16 + l15;
    if (col >= N) continue;
    const float bv = bias ? bias[col] : 0.0f;
#pragma unroll
    for (int i = 0; i < 2; ++i) {
      const int rbase = tm * 32 + i * 16 + half8;
#pragma unroll
      for (int r = 0; r < 8; ++r) {
        const int row = rbase + r;
        if (row >= M) continue;
        float vv = (acc[i][c][r] + bv) * post_scale;
        if (act == ACT_QUICKGELU) {
          vv = vv / (1.0f + __expf(-1.702f * vv));
        } else if (act == ACT_GELU) {
          float u = 0.7978845608028654f * (vv + 0.044715f * vv * vv * vv);
          vv = 0.5f * vv * (1.0f + tanhf(u));
        }
        if (res) vv += res[(size_t)row * N + col];
        if (out_mode == OUT_F32)       Cf[(size_t)row * N + col] = vv;
        else if (out_mode == OUT_BF16) Cb[(size_t)row * N + col] = (__bf16)vv;
        else                           Cb[(size_t)col * M + row] = (__bf16)vv;   // transposed
      }
    }
  }
}

// ---------------- layernorm: f32 in, f32 or bf16 out; pad rows -> zero ----------------
__global__ void __launch_bounds__(256)
layernorm_kernel(const float* __restrict__ x, const float* __restrict__ s,
                 const float* __restrict__ b, void* __restrict__ yout,
                 int out_bf16, int validRows, int D) {
  const int row = blockIdx.x;
  const int t = threadIdx.x;
  __shared__ float red[256];
  const float* xr = x + (size_t)row * D;
  float* yf = (float*)yout + (size_t)row * D;
  __bf16* yb = (__bf16*)yout + (size_t)row * D;
  if (row >= validRows) {
    for (int i = t; i < D; i += 256) { if (out_bf16) yb[i] = (__bf16)0.0f; else yf[i] = 0.0f; }
    return;
  }
  float sum = 0.f;
  for (int i = t; i < D; i += 256) sum += xr[i];
  red[t] = sum; __syncthreads();
  for (int o = 128; o > 0; o >>= 1) { if (t < o) red[t] += red[t + o]; __syncthreads(); }
  float mean = red[0] / D; __syncthreads();
  float vs = 0.f;
  for (int i = t; i < D; i += 256) { float d = xr[i] - mean; vs += d * d; }
  red[t] = vs; __syncthreads();
  for (int o = 128; o > 0; o >>= 1) { if (t < o) red[t] += red[t + o]; __syncthreads(); }
  float inv = rsqrtf(red[0] / D + 1e-5f);
  for (int i = t; i < D; i += 256) {
    float vv = (xr[i] - mean) * inv * s[i] + b[i];
    if (out_bf16) yb[i] = (__bf16)vv; else yf[i] = vv;
  }
}

// ---------------- flash attention ----------------
// One wave per (head, 16-query tile). Q tile staged global->LDS via async-to-LDS DMA
// (ASYNCcnt path); K fragments are contiguous b128 loads; V read from the transposed
// V^T [feature][token] buffer so P@V B-fragments are contiguous too.
__global__ void __launch_bounds__(256)
attention_kernel(const __bf16* __restrict__ q, const __bf16* __restrict__ k,
                 const __bf16* __restrict__ vT, __bf16* __restrict__ o) {
  __shared__ __bf16 sP[8][16 * 32];   // per-wave P staging (C-layout -> A-layout)
  __shared__ __bf16 sQ[8][16 * 64];   // per-wave Q tile (async DMA target)
  const int lane = threadIdx.x & 31;
  const int wave = threadIdx.x >> 5;
  const int QT = NTP / 16;
  const int task = blockIdx.x * 8 + wave;
  if (task >= NHEAD * QT) return;
  const int head = task / QT;
  const int qt = task - head * QT;
  const int hoff = head * HDIM;
  const int l15 = lane & 15;
  const int half8 = (lane & 16) >> 1;

  __bf16* myP = &sP[wave][0];
  __bf16* myQ = &sQ[wave][0];

  // ---- async DMA: copy 16x64 bf16 Q tile (2KB) into LDS, 16B per lane per issue ----
  {
    const unsigned lbase = (unsigned)(uintptr_t)myQ;          // low 32 bits = LDS address
    const char* gq = (const char*)(q + (size_t)(qt * 16) * DMODEL + hoff);
#pragma unroll
    for (int i = 0; i < 4; ++i) {
      const int off = i * 512 + lane * 16;
      const int grow = off >> 7;          // 128 bytes per tile row
      const int gcol = off & 127;
      const char* ga = gq + (size_t)grow * (DMODEL * 2) + gcol;
      const unsigned la = lbase + off;
      asm volatile("global_load_async_to_lds_b128 %0, %1, off"
                   :: "v"(la), "v"(ga) : "memory");
    }
    asm volatile("s_wait_asynccnt 0" ::: "memory");
  }
  v16bf a0 = frag_rowK_bf16(myQ, 64, l15, 0, lane);
  v16bf a1 = frag_rowK_bf16(myQ, 64, l15, 32, lane);

  v8f oacc[4];
#pragma unroll
  for (int c = 0; c < 4; ++c) oacc[c] = (v8f){};
  float m_run[8], l_run[8];
#pragma unroll
  for (int r = 0; r < 8; ++r) { m_run[r] = -3.0e38f; l_run[r] = 0.0f; }

  for (int kt = 0; kt < NTP / 32; ++kt) {
    const int kb = kt * 32;
    const int key0 = kb + l15;
    const int key1 = kb + 16 + l15;

    v8f s0 = {}, s1 = {};
    {
      v16bf b0 = frag_rowK_bf16(k, DMODEL, key0, hoff + 0, lane);
      v16bf b1 = frag_rowK_bf16(k, DMODEL, key0, hoff + 32, lane);
      s0 = __builtin_amdgcn_wmma_f32_16x16x32_bf16(false, a0, false, b0, (short)0, s0, false, false);
      s0 = __builtin_amdgcn_wmma_f32_16x16x32_bf16(false, a1, false, b1, (short)0, s0, false, false);
      v16bf c0 = frag_rowK_bf16(k, DMODEL, key1, hoff + 0, lane);
      v16bf c1 = frag_rowK_bf16(k, DMODEL, key1, hoff + 32, lane);
      s1 = __builtin_amdgcn_wmma_f32_16x16x32_bf16(false, a0, false, c0, (short)0, s1, false, false);
      s1 = __builtin_amdgcn_wmma_f32_16x16x32_bf16(false, a1, false, c1, (short)0, s1, false, false);
    }
    if (key0 >= LTOK) {
#pragma unroll
      for (int r = 0; r < 8; ++r) s0[r] = -3.0e30f;
    }
    if (key1 >= LTOK) {
#pragma unroll
      for (int r = 0; r < 8; ++r) s1[r] = -3.0e30f;
    }

    float alpha[8];
#pragma unroll
    for (int r = 0; r < 8; ++r) {
      // row reduction across the 16 lanes of each half-wave (rows r / r+8)
      float t = fmaxf(s0[r], s1[r]);
      t = fmaxf(t, __shfl_xor(t, 1, 32));
      t = fmaxf(t, __shfl_xor(t, 2, 32));
      t = fmaxf(t, __shfl_xor(t, 4, 32));
      t = fmaxf(t, __shfl_xor(t, 8, 32));
      float mnew = fmaxf(m_run[r], t);
      alpha[r] = __expf(m_run[r] - mnew);
      s0[r] = __expf(s0[r] - mnew);
      s1[r] = __expf(s1[r] - mnew);
      float rs = s0[r] + s1[r];
      rs += __shfl_xor(rs, 1, 32);
      rs += __shfl_xor(rs, 2, 32);
      rs += __shfl_xor(rs, 4, 32);
      rs += __shfl_xor(rs, 8, 32);
      l_run[r] = l_run[r] * alpha[r] + rs;
      m_run[r] = mnew;
    }
#pragma unroll
    for (int r = 0; r < 8; ++r) {
      oacc[0][r] *= alpha[r]; oacc[1][r] *= alpha[r];
      oacc[2][r] *= alpha[r]; oacc[3][r] *= alpha[r];
    }

    // P tile: C-layout -> LDS row-major 16x32 bf16
#pragma unroll
    for (int r = 0; r < 8; ++r) {
      const int prow = r + half8;
      myP[prow * 32 + l15]      = (__bf16)s0[r];
      myP[prow * 32 + 16 + l15] = (__bf16)s1[r];
    }
    asm volatile("s_wait_dscnt 0" ::: "memory");

    v16bf pf = frag_rowK_bf16(myP, 32, l15, 0, lane);

    // V^T fragments: row = feature column, K = token rows -> contiguous b128 loads
#pragma unroll
    for (int c = 0; c < 4; ++c) {
      v16bf vb = frag_rowK_bf16(vT, NTP, hoff + c * 16 + l15, kb, lane);
      oacc[c] = __builtin_amdgcn_wmma_f32_16x16x32_bf16(false, pf, false, vb, (short)0, oacc[c], false, false);
    }
  }

  float inv[8];
#pragma unroll
  for (int r = 0; r < 8; ++r) inv[r] = 1.0f / l_run[r];
#pragma unroll
  for (int r = 0; r < 8; ++r) {
    const int row = qt * 16 + r + half8;
    __bf16* orow = o + (size_t)row * DMODEL + hoff;
    orow[0 * 16 + l15] = (__bf16)(oacc[0][r] * inv[r]);
    orow[1 * 16 + l15] = (__bf16)(oacc[1][r] * inv[r]);
    orow[2 * 16 + l15] = (__bf16)(oacc[2][r] * inv[r]);
    orow[3 * 16 + l15] = (__bf16)(oacc[3][r] * inv[r]);
  }
}

// ---------------- elementwise / small kernels ----------------
__global__ void f32_to_bf16_kernel(const float* __restrict__ x, __bf16* __restrict__ y, long n) {
  long idx = (long)blockIdx.x * blockDim.x + threadIdx.x;
  if (idx >= n) return;
  y[idx] = (__bf16)x[idx];
}

__global__ void im2col_kernel(const float* __restrict__ pix, __bf16* __restrict__ X) {
  int idx = blockIdx.x * blockDim.x + threadIdx.x;
  if (idx >= NPATCH * DMODEL) return;
  int p = idx / DMODEL, col = idx % DMODEL;
  int c = col >> 8, rem = col & 255, ky = rem >> 4, kx = rem & 15;
  int gy = p / 48, gx = p % 48;
  const float mean[3] = {0.48145466f, 0.4578275f, 0.40821073f};
  const float stdv[3] = {0.26862954f, 0.26130258f, 0.27577711f};
  float val = pix[((size_t)c * 768 + (gy * 16 + ky)) * 768 + (gx * 16 + kx)];
  X[idx] = (__bf16)((val - mean[c]) / stdv[c]);
}

__global__ void build_h_kernel(const float* __restrict__ patches, const float* __restrict__ cls,
                               const float* __restrict__ pos, float* __restrict__ h) {
  int idx = blockIdx.x * blockDim.x + threadIdx.x;
  if (idx >= NTP * DMODEL) return;
  int row = idx / DMODEL, d = idx % DMODEL;
  float val;
  if (row == 0)           val = cls[d] + pos[d];
  else if (row < LTOK)    val = patches[(size_t)(row - 1) * DMODEL + d] + pos[(size_t)row * DMODEL + d];
  else                    val = 0.0f;
  h[idx] = val;
}

__global__ void merge_kernel(const __bf16* __restrict__ img, float* __restrict__ f) {
  int idx = blockIdx.x * blockDim.x + threadIdx.x;
  if (idx >= NPATCH * DMODEL) return;
  int p = idx / DMODEL, d = idx % DMODEL;
  f[idx] = (float)img[(size_t)(p + 1) * DMODEL + d] * (float)img[d];   // patch * cls token
}

__global__ void rownorm_f32_to_bf16_kernel(const float* __restrict__ x, __bf16* __restrict__ y,
                                           int rows, int D) {
  int r = blockIdx.x * blockDim.x + threadIdx.x;
  if (r >= rows) return;
  const float* p = x + (size_t)r * D;
  __bf16* qo = y + (size_t)r * D;
  float s = 0.f;
  for (int i = 0; i < D; ++i) { float v = p[i]; s += v * v; }
  float inv = 1.0f / (sqrtf(s) + 1e-6f);
  for (int i = 0; i < D; ++i) qo[i] = (__bf16)(p[i] * inv);
}

__global__ void shiftscale_kernel(const __bf16* __restrict__ feats,
                                  const float* __restrict__ sw, const float* __restrict__ sb,
                                  const float* __restrict__ cw, const float* __restrict__ cb,
                                  float* __restrict__ shiftv, float* __restrict__ scalev) {
  int r = blockIdx.x * blockDim.x + threadIdx.x;
  if (r >= NPATCH) return;
  const __bf16* f = feats + (size_t)r * DMODEL;
  float s1 = 0.f, s2 = 0.f;
  for (int i = 0; i < DMODEL; ++i) { float fi = (float)f[i]; s1 += fi * sw[i]; s2 += fi * cw[i]; }
  shiftv[r] = s1 + sb[0];
  float e = s2 + cb[0];
  scalev[r] = (e > 0.f ? e : (__expf(e) - 1.0f)) + 1.0f;   // elu + 1
}

__global__ void logits_post_kernel(float* __restrict__ L, const float* __restrict__ shiftv,
                                   const float* __restrict__ scalev,
                                   const unsigned char* __restrict__ mask) {
  int idx = blockIdx.x * blockDim.x + threadIdx.x;
  if (idx >= NPATCH * NQ) return;
  int p = idx / NQ, qq = idx - p * NQ;
  float val = (L[idx] + shiftv[p]) * scalev[p];
  if (!mask[qq]) val = -1e6f;
  L[idx] = val;
}

__global__ void boxes_kernel(const float* __restrict__ b3, const float* __restrict__ bias,
                             float* __restrict__ out) {
  int idx = blockIdx.x * blockDim.x + threadIdx.x;
  if (idx >= NPATCH * 4) return;
  float v = b3[idx] + bias[idx];
  out[idx] = 1.0f / (1.0f + __expf(-v));
}

__global__ void scores_kernel(const float* __restrict__ L, float* __restrict__ scores,
                              int* __restrict__ labels, int* __restrict__ validv) {
  int p = blockIdx.x * blockDim.x + threadIdx.x;
  if (p >= NPATCH) return;
  const float* row = L + (size_t)p * NQ;
  float best = -3.0e38f; int bi = 0;
  for (int qq = 0; qq < NQ; ++qq) { float v = row[qq]; if (v > best) { best = v; bi = qq; } }
  float sc = 1.0f / (1.0f + __expf(-best));   // sigmoid monotone: max/argmax commute
  scores[p] = sc; labels[p] = bi; validv[p] = (sc > 0.2f) ? 1 : 0;
}

__global__ void nmsprep_kernel(const float* __restrict__ boxes, const int* __restrict__ labels,
                               float* __restrict__ bb, float* __restrict__ area) {
  int p = blockIdx.x * blockDim.x + threadIdx.x;
  if (p >= NPATCH) return;
  float off = 4.0f * (float)labels[p];
  float x1 = boxes[p * 4 + 0] + off, y1 = boxes[p * 4 + 1] + off;
  float x2 = boxes[p * 4 + 2] + off, y2 = boxes[p * 4 + 3] + off;
  bb[p * 4 + 0] = x1; bb[p * 4 + 1] = y1; bb[p * 4 + 2] = x2; bb[p * 4 + 3] = y2;
  area[p] = (x2 - x1) * (y2 - y1);
}

// rank-sort == argsort(-masked_score) with stable tie-break by original index
__global__ void sort_kernel(const float* __restrict__ scores, const int* __restrict__ validv,
                            const float* __restrict__ bb, const float* __restrict__ area,
                            float* __restrict__ sb, float* __restrict__ sarea,
                            int* __restrict__ svalid, int* __restrict__ sidx) {
  int i = blockIdx.x * blockDim.x + threadIdx.x;
  if (i >= NPATCH) return;
  float si = validv[i] ? scores[i] : -1.0f;
  int rank = 0;
  for (int j = 0; j < NPATCH; ++j) {
    float sj = validv[j] ? scores[j] : -1.0f;
    rank += (sj > si) || (sj == si && j < i);
  }
  sb[rank * 4 + 0] = bb[i * 4 + 0]; sb[rank * 4 + 1] = bb[i * 4 + 1];
  sb[rank * 4 + 2] = bb[i * 4 + 2]; sb[rank * 4 + 3] = bb[i * 4 + 3];
  sarea[rank] = area[i]; svalid[rank] = validv[i]; sidx[rank] = i;
}

__global__ void __launch_bounds__(256)
nms_kernel(const float* __restrict__ sb, const float* __restrict__ sarea,
           const int* __restrict__ svalid, const int* __restrict__ sidx,
           int* __restrict__ supp, float* __restrict__ keep) {
  const int t = threadIdx.x;
  for (int i = t; i < NPATCH; i += 256) supp[i] = 0;
  __shared__ float cb0, cb1, cb2, cb3, carea;
  __shared__ int ckeep;
  __syncthreads();
  for (int i = 0; i < NPATCH; ++i) {
    if (t == 0) {
      int kk = (supp[i] == 0) && svalid[i];
      ckeep = kk;
      cb0 = sb[i * 4 + 0]; cb1 = sb[i * 4 + 1];
      cb2 = sb[i * 4 + 2]; cb3 = sb[i * 4 + 3];
      carea = sarea[i];
      keep[sidx[i]] = kk ? 1.0f : 0.0f;
    }
    __syncthreads();
    if (ckeep) {
      for (int j = i + 1 + t; j < NPATCH; j += 256) {
        float ix1 = fmaxf(cb0, sb[j * 4 + 0]);
        float iy1 = fmaxf(cb1, sb[j * 4 + 1]);
        float ix2 = fminf(cb2, sb[j * 4 + 2]);
        float iy2 = fminf(cb3, sb[j * 4 + 3]);
        float inter = fmaxf(ix2 - ix1, 0.f) * fmaxf(iy2 - iy1, 0.f);
        float iou = inter / (carea + sarea[j] - inter + 1e-6f);
        if (iou > 0.5f) supp[j] = 1;
      }
    }
    __syncthreads();
  }
}

// ---------------- host orchestration ----------------
extern "C" void kernel_launch(void* const* d_in, const int* in_sizes, int n_in,
                              void* d_out, int out_size, void* d_ws, size_t ws_size,
                              hipStream_t stream) {
  (void)in_sizes; (void)n_in; (void)out_size; (void)ws_size;
  const float* pixel        = (const float*)d_in[0];
  const float* query_embeds = (const float*)d_in[1];
  const unsigned char* query_mask = (const unsigned char*)d_in[2];
  int ii = 3;
  const float* box1_b = (const float*)d_in[ii++];
  const float* box1_w = (const float*)d_in[ii++];
  const float* box2_b = (const float*)d_in[ii++];
  const float* box2_w = (const float*)d_in[ii++];
  const float* box3_b = (const float*)d_in[ii++];
  const float* box3_w = (const float*)d_in[ii++];
  const float* box_bias = (const float*)d_in[ii++];
  const float* class_embedding = (const float*)d_in[ii++];
  const float* cls_dense0_b = (const float*)d_in[ii++];
  const float* cls_dense0_w = (const float*)d_in[ii++];
  const float* fc1_b = (const float*)d_in[ii++];
  const float* fc1_w = (const float*)d_in[ii++];
  const float* fc2_b = (const float*)d_in[ii++];
  const float* fc2_w = (const float*)d_in[ii++];
  const float* k_b = (const float*)d_in[ii++];
  const float* k_w = (const float*)d_in[ii++];
  const float* ln1_b = (const float*)d_in[ii++];
  const float* ln1_s = (const float*)d_in[ii++];
  const float* ln2_b = (const float*)d_in[ii++];
  const float* ln2_s = (const float*)d_in[ii++];
  const float* o_b = (const float*)d_in[ii++];
  const float* o_w = (const float*)d_in[ii++];
  const float* q_b = (const float*)d_in[ii++];
  const float* q_w = (const float*)d_in[ii++];
  const float* v_b = (const float*)d_in[ii++];
  const float* v_w = (const float*)d_in[ii++];
  const float* merge_ln_b = (const float*)d_in[ii++];
  const float* merge_ln_s = (const float*)d_in[ii++];
  const float* patch_kernel = (const float*)d_in[ii++];
  const float* pos_embedding = (const float*)d_in[ii++];
  const float* post_ln_b = (const float*)d_in[ii++];
  const float* post_ln_s = (const float*)d_in[ii++];
  const float* pre_ln_b = (const float*)d_in[ii++];
  const float* pre_ln_s = (const float*)d_in[ii++];
  const float* scale_b = (const float*)d_in[ii++];
  const float* scale_w = (const float*)d_in[ii++];
  const float* shift_b = (const float*)d_in[ii++];
  const float* shift_w = (const float*)d_in[ii++];

  float* out_logits = (float*)d_out;                         // [2304,1203]
  float* out_boxes  = out_logits + (size_t)NPATCH * NQ;      // [2304,4]
  float* out_keep   = out_boxes + (size_t)NPATCH * 4;        // [2304]

  // ---- workspace layout (byte cursor) ----
  char* wsb = (char*)d_ws;
  auto allocB = [&](size_t bytes) { char* p = wsb; wsb += (bytes + 63) & ~(size_t)63; return p; };
  auto allocF = [&](size_t n) { return (float*)allocB(n * 4); };
  auto allocH = [&](size_t n) { return (__bf16*)allocB(n * 2); };

  // bf16 weight cache (~174 MB -> fits the 192 MB global L2; re-reads across M-tiles hit L2)
  __bf16* patchW = allocH((size_t)DMODEL * DMODEL);
  __bf16* qW  = allocH((size_t)NLAYER * DMODEL * DMODEL);
  __bf16* kW  = allocH((size_t)NLAYER * DMODEL * DMODEL);
  __bf16* vW  = allocH((size_t)NLAYER * DMODEL * DMODEL);
  __bf16* oW  = allocH((size_t)NLAYER * DMODEL * DMODEL);
  __bf16* f1W = allocH((size_t)NLAYER * IMLP * DMODEL);
  __bf16* f2W = allocH((size_t)NLAYER * DMODEL * IMLP);
  __bf16* clsW  = allocH((size_t)QD * DMODEL);
  __bf16* b1W = allocH((size_t)DMODEL * DMODEL);
  __bf16* b2W = allocH((size_t)DMODEL * DMODEL);
  __bf16* b3W = allocH((size_t)4 * DMODEL);
  // bf16 activations
  __bf16* X   = allocH((size_t)NPATCH * DMODEL);
  __bf16* ybf = allocH((size_t)NTP * DMODEL);
  __bf16* qb  = allocH((size_t)NTP * DMODEL);
  __bf16* kb  = allocH((size_t)NTP * DMODEL);
  __bf16* vTb = allocH((size_t)DMODEL * NTP);   // V^T [feature][token]
  __bf16* ob  = allocH((size_t)NTP * DMODEL);
  __bf16* mb  = allocH((size_t)NTP * IMLP);
  __bf16* fb2b = allocH((size_t)NPATCH * DMODEL);
  __bf16* icen = allocH((size_t)NPATCH * QD);
  __bf16* qen  = allocH((size_t)NQ * QD);
  // f32 buffers
  float* hraw = allocF((size_t)NTP * DMODEL);
  float* hb   = allocF((size_t)NTP * DMODEL);     // fp32 residual stream
  float* Pm   = allocF((size_t)NPATCH * DMODEL);
  float* fb   = allocF((size_t)NPATCH * DMODEL);
  float* icef = allocF((size_t)NPATCH * QD);
  float* shiftv = allocF(NPATCH);
  float* scalev = allocF(NPATCH);
  float* b3buf  = allocF((size_t)NPATCH * 4);
  float* scoresv = allocF(NPATCH);
  int*   labels  = (int*)allocF(NPATCH);
  int*   validv  = (int*)allocF(NPATCH);
  float* bbv     = allocF((size_t)NPATCH * 4);
  float* areav   = allocF(NPATCH);
  float* sbv     = allocF((size_t)NPATCH * 4);
  float* sareav  = allocF(NPATCH);
  int*   svalid  = (int*)allocF(NPATCH);
  int*   sidx    = (int*)allocF(NPATCH);
  int*   supp    = (int*)allocF(NPATCH);

  auto ew = [&](long n) { return (int)((n + 255) / 256); };
  auto conv = [&](const float* src, __bf16* dst, long n) {
    f32_to_bf16_kernel<<<ew(n), 256, 0, stream>>>(src, dst, n);
  };
  auto gemm = [&](const __bf16* A, const __bf16* W, const float* bias, const float* res,
                  void* C, int M, int N, int K, float scale, int act, int outm) {
    int tiles = (M >> 5) * ceil_div_i(N, 64);
    gemm_wmma_kernel<<<ceil_div_i(tiles, 8), 256, 0, stream>>>(A, W, bias, res, C, M, N, K, scale, act, outm);
  };

  // ---- narrow all weights to bf16 once (deterministic each call) ----
  conv(patch_kernel, patchW, (long)DMODEL * DMODEL);
  conv(q_w, qW, (long)NLAYER * DMODEL * DMODEL);
  conv(k_w, kW, (long)NLAYER * DMODEL * DMODEL);
  conv(v_w, vW, (long)NLAYER * DMODEL * DMODEL);
  conv(o_w, oW, (long)NLAYER * DMODEL * DMODEL);
  conv(fc1_w, f1W, (long)NLAYER * IMLP * DMODEL);
  conv(fc2_w, f2W, (long)NLAYER * DMODEL * IMLP);
  conv(cls_dense0_w, clsW, (long)QD * DMODEL);
  conv(box1_w, b1W, (long)DMODEL * DMODEL);
  conv(box2_w, b2W, (long)DMODEL * DMODEL);
  conv(box3_w, b3W, (long)4 * DMODEL);

  // ---- patch embedding ----
  im2col_kernel<<<ew(NPATCH * DMODEL), 256, 0, stream>>>(pixel, X);
  gemm(X, patchW, nullptr, nullptr, Pm, NPATCH, DMODEL, DMODEL, 1.0f, ACT_NONE, OUT_F32);
  build_h_kernel<<<ew(NTP * DMODEL), 256, 0, stream>>>(Pm, class_embedding, pos_embedding, hraw);
  layernorm_kernel<<<NTP, 256, 0, stream>>>(hraw, pre_ln_s, pre_ln_b, hb, 0, LTOK, DMODEL);

  // ---- transformer encoder ----
  const float attn_scale = 0.125f; // 1/sqrt(64)
  for (int l = 0; l < NLAYER; ++l) {
    const size_t dd = (size_t)l * DMODEL * DMODEL;
    const size_t d1 = (size_t)l * DMODEL;
    const size_t di = (size_t)l * IMLP * DMODEL;
    layernorm_kernel<<<NTP, 256, 0, stream>>>(hb, ln1_s + d1, ln1_b + d1, ybf, 1, LTOK, DMODEL);
    gemm(ybf, qW + dd, q_b + d1, nullptr, qb, NTP, DMODEL, DMODEL, attn_scale, ACT_NONE, OUT_BF16);
    gemm(ybf, kW + dd, k_b + d1, nullptr, kb, NTP, DMODEL, DMODEL, 1.0f, ACT_NONE, OUT_BF16);
    gemm(ybf, vW + dd, v_b + d1, nullptr, vTb, NTP, DMODEL, DMODEL, 1.0f, ACT_NONE, OUT_BF16T);
    attention_kernel<<<ceil_div_i(NHEAD * (NTP / 16), 8), 256, 0, stream>>>(qb, kb, vTb, ob);
    gemm(ob, oW + dd, o_b + d1, hb, hb, NTP, DMODEL, DMODEL, 1.0f, ACT_NONE, OUT_F32);
    layernorm_kernel<<<NTP, 256, 0, stream>>>(hb, ln2_s + d1, ln2_b + d1, ybf, 1, LTOK, DMODEL);
    gemm(ybf, f1W + di, fc1_b + (size_t)l * IMLP, nullptr, mb, NTP, IMLP, DMODEL, 1.0f, ACT_QUICKGELU, OUT_BF16);
    gemm(mb, f2W + di, fc2_b + d1, hb, hb, NTP, DMODEL, IMLP, 1.0f, ACT_NONE, OUT_F32);
  }

  // ---- post LN + merge class token ----
  layernorm_kernel<<<NTP, 256, 0, stream>>>(hb, post_ln_s, post_ln_b, ybf, 1, LTOK, DMODEL);
  merge_kernel<<<ew(NPATCH * DMODEL), 256, 0, stream>>>(ybf, fb);
  layernorm_kernel<<<NPATCH, 256, 0, stream>>>(fb, merge_ln_s, merge_ln_b, fb2b, 1, NPATCH, DMODEL);

  // ---- class head ----
  gemm(fb2b, clsW, cls_dense0_b, nullptr, icef, NPATCH, QD, DMODEL, 1.0f, ACT_NONE, OUT_F32);
  rownorm_f32_to_bf16_kernel<<<ew(NPATCH), 256, 0, stream>>>(icef, icen, NPATCH, QD);
  rownorm_f32_to_bf16_kernel<<<ew(NQ), 256, 0, stream>>>(query_embeds, qen, NQ, QD);
  gemm(icen, qen, nullptr, nullptr, out_logits, NPATCH, NQ, QD, 1.0f, ACT_NONE, OUT_F32);
  shiftscale_kernel<<<ew(NPATCH), 256, 0, stream>>>(fb2b, shift_w, shift_b, scale_w, scale_b, shiftv, scalev);
  logits_post_kernel<<<ew((long)NPATCH * NQ), 256, 0, stream>>>(out_logits, shiftv, scalev, query_mask);

  // ---- box head ----
  gemm(fb2b, b1W, box1_b, nullptr, qb, NPATCH, DMODEL, DMODEL, 1.0f, ACT_GELU, OUT_BF16);
  gemm(qb, b2W, box2_b, nullptr, kb, NPATCH, DMODEL, DMODEL, 1.0f, ACT_GELU, OUT_BF16);
  gemm(kb, b3W, box3_b, nullptr, b3buf, NPATCH, 4, DMODEL, 1.0f, ACT_NONE, OUT_F32);
  boxes_kernel<<<ew(NPATCH * 4), 256, 0, stream>>>(b3buf, box_bias, out_boxes);

  // ---- scores + NMS ----
  scores_kernel<<<ew(NPATCH), 256, 0, stream>>>(out_logits, scoresv, labels, validv);
  nmsprep_kernel<<<ew(NPATCH), 256, 0, stream>>>(out_boxes, labels, bbv, areav);
  sort_kernel<<<ew(NPATCH), 256, 0, stream>>>(scoresv, validv, bbv, areav, sbv, sareav, svalid, sidx);
  nms_kernel<<<1, 256, 0, stream>>>(sbv, sareav, svalid, sidx, supp, out_keep);
}
